// V4_SemanticAbstraction_85667417686648
// MI455X (gfx1250) — matrix operand order repaired and checked by reading, not compiled, not run on hardware
//
#include <hip/hip_runtime.h>

// ---------------------------------------------------------------------------
// Mamba forward for gfx1250 (MI455X): bf16 WMMA GEMMs + fused LDS-chunked scan
// ---------------------------------------------------------------------------

typedef __attribute__((ext_vector_type(16))) __bf16 v16bf;
typedef __attribute__((ext_vector_type(8)))  float  v8f;
typedef __bf16 bf16;

#define B_    4
#define DIM_  256
#define DI_   512   // d_inner
#define NST_  16    // d_state
#define L_    4096  // h*w
#define DTR_  16    // dt_rank

struct __align__(16) U4 { unsigned x, y, z, w; };
union AFrag { v16bf v; U4 u[2]; };
struct __align__(8) BF4 { bf16 v[4]; };

__device__ __forceinline__ bf16 f2bf(float f) {
  unsigned u = __builtin_bit_cast(unsigned, f);
  unsigned r = (u + 0x7FFFu + ((u >> 16) & 1u)) >> 16;
  return __builtin_bit_cast(bf16, (unsigned short)r);
}

__device__ __forceinline__ float silu(float x) {
  return x / (1.f + __expf(-x));
}

// ------------------------- f32 -> bf16 conversion --------------------------
__global__ void cvt_f32_bf16(const float* __restrict__ in, bf16* __restrict__ out, int n) {
  int i = blockIdx.x * blockDim.x + threadIdx.x;
  if (i < n) out[i] = f2bf(in[i]);
}

// ---------------------------------------------------------------------------
// WMMA GEMM:  Out[b][m][l] = sum_k W[m][k] * Act[b][k][l]
//   W   : bf16 row-major [M x K]          (K compile-time, multiple of 32)
//   Act : bf16 [K x L_] per batch, l contiguous
//   Out : f32  [M x L_] per batch
// block = 128 threads (4 waves). Each wave: 32(M) x 64(L) tile,
// software-pipelined (double-buffered fragments), fully unrolled K loop.
// grid = (L_/256, ceil(M/32), B_)
// Fragment layouts (CDNA5 ISA 7.12.2):
//   A (16x32 bf16): lane m<16 -> row m, K = {kA..kA+7, kA+16..kA+23}, kA=(lane<16?0:8)
//       -> two 8-elem chunks at +0 and +16 elements
//   B (32x16 bf16): lane n<16 -> col n, K = {kB..kB+7, kB+16..kB+23}, kB=(lane<16?0:8)
//       -> strided gather (stride L_) over K
//   D (16x16 f32) : VGPR r -> row r (lanes 0-15) / row r+8 (lanes 16-31), N = lane&15
// ---------------------------------------------------------------------------
__device__ __forceinline__ void loadA(AFrag& f, const bf16* p) {
  f.u[0] = *(const U4*)(p);        // K = kA .. kA+7
  f.u[1] = *(const U4*)(p + 16);   // K = kA+16 .. kA+23
}
__device__ __forceinline__ void loadB(AFrag& f, const bf16* p) {
#pragma unroll
  for (int s = 0; s < 8; ++s) {
    f.v[s]     = p[(long long)s * L_];          // K = kB+s
    f.v[8 + s] = p[(long long)(16 + s) * L_];   // K = kB+16+s
  }
}

template <int K>
__global__ void gemm_bf16_wmma(const bf16* __restrict__ W, const bf16* __restrict__ Act,
                               float* __restrict__ Out, int M,
                               long long actStride, long long outStride) {
  const int b    = blockIdx.z;
  const int m0   = blockIdx.y * 32;
  const int wave = threadIdx.x >> 5;
  const int lane = threadIdx.x & 31;
  const int n0   = blockIdx.x * 256 + wave * 64;

  const bf16* Ab = Act + (long long)b * actStride;
  float*      Ob = Out + (long long)b * outStride;

  const int mA = lane & 15;
  const int kA = (lane < 16) ? 0 : 8;
  const int kB = (lane < 16) ? 0 : 8;

  const bool hasM2 = (m0 + 16) < M;
  const bf16* arow0 = W + (long long)(m0 + mA) * K;
  const bf16* arow1 = W + (long long)((hasM2 ? m0 + 16 : m0) + mA) * K;

  v8f acc0[4], acc1[4];
#pragma unroll
  for (int t = 0; t < 4; ++t) { acc0[t] = v8f{}; acc1[t] = v8f{}; }

  // --- prologue: fragments for k0 = 0 ---
  AFrag a0, a1, bb[4];
  loadA(a0, arow0 + kA);
  loadA(a1, arow1 + kA);
  {
    const bf16* brow = Ab + (long long)kB * L_ + n0 + (lane & 15);
#pragma unroll
    for (int t = 0; t < 4; ++t) loadB(bb[t], brow + t * 16);
  }

#pragma unroll
  for (int k0 = 0; k0 < K; k0 += 32) {
    AFrag na0, na1, nb[4];
    constexpr int STEP = 32;
    const bool notLast = (k0 + STEP) < K;      // compile-time after unroll
    if (notLast) {
      loadA(na0, arow0 + k0 + STEP + kA);
      loadA(na1, arow1 + k0 + STEP + kA);
      const bf16* brow = Ab + (long long)(k0 + STEP + kB) * L_ + n0 + (lane & 15);
#pragma unroll
      for (int t = 0; t < 4; ++t) loadB(nb[t], brow + t * 16);
    }
#pragma unroll
    for (int t = 0; t < 4; ++t) {
      acc0[t] = __builtin_amdgcn_wmma_f32_16x16x32_bf16(
          false, a0.v, false, bb[t].v, (short)0, acc0[t], false, false);
      acc1[t] = __builtin_amdgcn_wmma_f32_16x16x32_bf16(
          false, a1.v, false, bb[t].v, (short)0, acc1[t], false, false);
    }
    if (notLast) {
      a0 = na0;
      a1 = na1;
#pragma unroll
      for (int t = 0; t < 4; ++t) bb[t] = nb[t];
    }
  }

  const int rowOff = (lane < 16) ? 0 : 8;
  const int col    = lane & 15;
#pragma unroll
  for (int t = 0; t < 4; ++t)
#pragma unroll
    for (int r = 0; r < 8; ++r)
      Ob[(long long)(m0 + rowOff + r) * L_ + n0 + t * 16 + col] = acc0[t][r];
  if (hasM2) {
#pragma unroll
    for (int t = 0; t < 4; ++t)
#pragma unroll
      for (int r = 0; r < 8; ++r)
        Ob[(long long)(m0 + 16 + rowOff + r) * L_ + n0 + t * 16 + col] = acc1[t][r];
  }
}

// ------------------- causal depthwise conv(k=4) + SiLU ---------------------
// xz layout: [B][2*DI][L]; rows 0..511 = u_pre, 512..1023 = res
__global__ void conv_silu_kernel(const float* __restrict__ xz,
                                 const float* __restrict__ conv_w,
                                 const float* __restrict__ conv_b,
                                 float* __restrict__ u, bf16* __restrict__ u_bf) {
  int idx = blockIdx.x * blockDim.x + threadIdx.x;   // B*DI*L threads
  int l = idx & (L_ - 1);
  int d = (idx >> 12) & (DI_ - 1);
  int b = idx >> 21;
  const float* row = xz + ((long long)b * (2 * DI_) + d) * L_;
  const float* w   = conv_w + d * 4;
  float acc = conv_b[d];
#pragma unroll
  for (int j = 0; j < 4; ++j) {
    int li = l - 3 + j;
    float v = (li >= 0) ? row[li] : 0.f;
    acc += w[j] * v;
  }
  float s = silu(acc);
  long long o = ((long long)b * DI_ + d) * L_ + l;
  u[o]    = s;
  u_bf[o] = f2bf(s);
}

// ----------------------- dt_proj (K=16) + softplus --------------------------
// dbc layout: [B][48][L]; rows 0..15 = dt, 16..31 = Bm, 32..47 = Cm
__global__ void delta_kernel(const float* __restrict__ dbc,
                             const float* __restrict__ Wdt,   // [DI][16]
                             const float* __restrict__ bdt,
                             float* __restrict__ delta) {
  int idx = blockIdx.x * blockDim.x + threadIdx.x;   // B*DI*L threads
  int l = idx & (L_ - 1);
  int d = (idx >> 12) & (DI_ - 1);
  int b = idx >> 21;
  const float* base = dbc + (long long)b * 48 * L_ + l;
  float acc = bdt[d];
#pragma unroll
  for (int r = 0; r < DTR_; ++r)
    acc += Wdt[d * DTR_ + r] * base[(long long)r * L_];
  float sp = (acc > 20.f) ? acc : __logf(1.f + __expf(acc));
  delta[((long long)b * DI_ + d) * L_ + l] = sp;
}

// ------------------------------ selective scan ------------------------------
// grid = B_, block = DI_ (one channel per thread, state[16] in registers).
// B/C vectors staged in LDS in 64-step chunks; fuses +u*D, *silu(res), bf16 cvt.
#define CHUNK_ 64
__global__ void scan_kernel(const float* __restrict__ delta,
                            const float* __restrict__ u,
                            const float* __restrict__ dbc,
                            const float* __restrict__ xz,     // res rows
                            const float* __restrict__ A_log,
                            const float* __restrict__ Dvec,
                            bf16* __restrict__ yact) {
  __shared__ float Bs[NST_][CHUNK_];
  __shared__ float Cs[NST_][CHUNK_];

  const int b = blockIdx.x;
  const int d = threadIdx.x;

  float A_d[NST_];
#pragma unroll
  for (int n = 0; n < NST_; ++n) A_d[n] = -__expf(A_log[d * NST_ + n]);
  const float Dd = Dvec[d];

  float state[NST_];
#pragma unroll
  for (int n = 0; n < NST_; ++n) state[n] = 0.f;

  const float* drow = delta + ((long long)b * DI_ + d) * L_;
  const float* urow = u     + ((long long)b * DI_ + d) * L_;
  const float* rrow = xz    + ((long long)b * (2 * DI_) + DI_ + d) * L_;
  bf16*        yrow = yact  + ((long long)b * DI_ + d) * L_;
  const float* dbcb = dbc   + (long long)b * 48 * L_;

  for (int l0 = 0; l0 < L_; l0 += CHUNK_) {
    // cooperative load of B/C chunk: 16*64 floats each, 512 threads -> 2 per thread
#pragma unroll
    for (int i = 0; i < 2; ++i) {
      int e = threadIdx.x + i * DI_;
      int n = e >> 6;
      int j = e & (CHUNK_ - 1);
      Bs[n][j] = dbcb[(long long)(16 + n) * L_ + l0 + j];
      Cs[n][j] = dbcb[(long long)(32 + n) * L_ + l0 + j];
    }
    __syncthreads();

    for (int jb = 0; jb < CHUNK_; jb += 4) {
      float4 dv = *(const float4*)(drow + l0 + jb);
      float4 uv = *(const float4*)(urow + l0 + jb);
      float4 rv = *(const float4*)(rrow + l0 + jb);
      float da[4] = {dv.x, dv.y, dv.z, dv.w};
      float ua[4] = {uv.x, uv.y, uv.z, uv.w};
      float ra[4] = {rv.x, rv.y, rv.z, rv.w};
      BF4 pack;
#pragma unroll
      for (int q = 0; q < 4; ++q) {
        const int j = jb + q;
        const float dt = da[q];
        const float uu = ua[q];
        float y = 0.f;
#pragma unroll
        for (int n = 0; n < NST_; ++n) {
          float dA = __expf(dt * A_d[n]);
          float s  = dA * state[n] + dt * Bs[n][j] * uu;
          state[n] = s;
          y += s * Cs[n][j];
        }
        y += uu * Dd;
        pack.v[q] = f2bf(y * silu(ra[q]));
      }
      *(BF4*)(yrow + l0 + jb) = pack;
    }
    __syncthreads();
  }
}

// ---------------------------------------------------------------------------
extern "C" void kernel_launch(void* const* d_in, const int* in_sizes, int n_in,
                              void* d_out, int out_size, void* d_ws, size_t ws_size,
                              hipStream_t stream) {
  const float* x       = (const float*)d_in[0];   // [B,256,64,64]
  const float* in_w    = (const float*)d_in[1];   // [1024,256]
  const float* conv_w  = (const float*)d_in[2];   // [512,1,4]
  const float* conv_b  = (const float*)d_in[3];   // [512]
  const float* xproj_w = (const float*)d_in[4];   // [48,512]
  const float* dt_w    = (const float*)d_in[5];   // [512,16]
  const float* dt_b    = (const float*)d_in[6];   // [512]
  const float* A_log   = (const float*)d_in[7];   // [512,16]
  const float* Dvec    = (const float*)d_in[8];   // [512]
  const float* out_w   = (const float*)d_in[9];   // [256,512]
  float* out = (float*)d_out;

  char* ws = (char*)d_ws;
  size_t off = 0;
  auto alloc = [&](size_t bytes) {
    void* p = ws + off;
    off = (off + bytes + 255) & ~(size_t)255;
    return p;
  };

  float* xz      = (float*)alloc((size_t)B_ * 2 * DI_ * L_ * 4);   // in_proj out
  bf16*  x_bf    = (bf16*) alloc((size_t)B_ * DIM_ * L_ * 2);
  bf16*  win_bf  = (bf16*) alloc((size_t)2 * DI_ * DIM_ * 2);
  bf16*  wxp_bf  = (bf16*) alloc((size_t)48 * DI_ * 2);
  bf16*  wout_bf = (bf16*) alloc((size_t)DIM_ * DI_ * 2);
  float* u       = (float*)alloc((size_t)B_ * DI_ * L_ * 4);
  bf16*  u_bf    = (bf16*) alloc((size_t)B_ * DI_ * L_ * 2);
  float* dbc     = (float*)alloc((size_t)B_ * 48 * L_ * 4);
  float* delta   = (float*)alloc((size_t)B_ * DI_ * L_ * 4);
  bf16*  yact_bf = (bf16*) alloc((size_t)B_ * DI_ * L_ * 2);

  // --- weight / activation conversions to bf16 ---
  {
    int n;
    n = B_ * DIM_ * L_;
    cvt_f32_bf16<<<(n + 255) / 256, 256, 0, stream>>>(x, x_bf, n);
    n = 2 * DI_ * DIM_;
    cvt_f32_bf16<<<(n + 255) / 256, 256, 0, stream>>>(in_w, win_bf, n);
    n = 48 * DI_;
    cvt_f32_bf16<<<(n + 255) / 256, 256, 0, stream>>>(xproj_w, wxp_bf, n);
    n = DIM_ * DI_;
    cvt_f32_bf16<<<(n + 255) / 256, 256, 0, stream>>>(out_w, wout_bf, n);
  }

  // --- in_proj: xz[b][e][l] = sum_d Win[e][d] * x[b][d][l] ---
  gemm_bf16_wmma<DIM_><<<dim3(L_ / 256, (2 * DI_) / 32, B_), 128, 0, stream>>>(
      win_bf, x_bf, xz, 2 * DI_,
      (long long)DIM_ * L_, (long long)2 * DI_ * L_);

  // --- causal conv + SiLU on u half ---
  {
    int n = B_ * DI_ * L_;
    conv_silu_kernel<<<n / 256, 256, 0, stream>>>(xz, conv_w, conv_b, u, u_bf);
  }

  // --- x_proj: dbc[b][j][l] = sum_e Wxp[j][e] * u[b][e][l] ---  (M=48 -> 2 tiles)
  gemm_bf16_wmma<DI_><<<dim3(L_ / 256, (48 + 31) / 32, B_), 128, 0, stream>>>(
      wxp_bf, u_bf, dbc, 48,
      (long long)DI_ * L_, (long long)48 * L_);

  // --- dt_proj + softplus ---
  {
    int n = B_ * DI_ * L_;
    delta_kernel<<<n / 256, 256, 0, stream>>>(dbc, dt_w, dt_b, delta);
  }

  // --- selective scan (fused +u*D, *silu(res), bf16 cvt) ---
  scan_kernel<<<B_, DI_, 0, stream>>>(delta, u, dbc, xz, A_log, Dvec, yact_bf);

  // --- out_proj: out[b][c][l] = sum_e Wout[c][e] * yact[b][e][l] ---
  gemm_bf16_wmma<DI_><<<dim3(L_ / 256, DIM_ / 32, B_), 128, 0, stream>>>(
      wout_bf, yact_bf, out, DIM_,
      (long long)DI_ * L_, (long long)DIM_ * L_);
}
